// TXCDRCausal_90984587198483
// MI455X (gfx1250) — matrix operand order repaired and checked by reading, not compiled, not run on hardware
//
#include <hip/hip_runtime.h>
#include <hip/hip_bf16.h>

// Problem constants (from reference)
#define Bn    8
#define Tn    32
#define D_IN  768
#define D_SAE 8192
#define TOPK  64
#define NROWS (Bn * Tn)        // 256 (b,t) rows
#define NBLK  64               // columns of D_SAE per workgroup -> 128 WGs
#define KPAD  784              // 768 + 16 bf16 pad -> 32B-aligned column stride
#define XROWS 64               // padded time rows per b: [0..31]=zeros, [32..63]=x[b][0..31]

typedef __attribute__((ext_vector_type(2)))  __bf16 v2bf;
typedef __attribute__((ext_vector_type(4)))  float  v4f;
typedef __attribute__((ext_vector_type(8)))  float  v8f;
typedef __attribute__((ext_vector_type(8)))  __bf16 v8bf;
typedef __attribute__((ext_vector_type(16))) __bf16 v16bf;

// ---------------------------------------------------------------------------
// Kernel 0: build zero-padded bf16 X in workspace.
//   Xpad[b][r][k], r in [0,64): r<32 -> 0, else x[b][r-32][k].
// ---------------------------------------------------------------------------
__global__ __launch_bounds__(256) void sae_build_xpad(
    const float* __restrict__ x, __bf16* __restrict__ xp)
{
    const int base = (blockIdx.x * 256 + threadIdx.x) * 4;
    #pragma unroll
    for (int e = 0; e < 4; ++e) {
        int idx = base + e;                      // < 8*64*768
        int k = idx % D_IN;
        int r = (idx / D_IN) & (XROWS - 1);
        int b = idx / (D_IN * XROWS);
        xp[idx] = (r < 32) ? (__bf16)0.0f
                           : (__bf16)x[((size_t)(b * Tn + (r - 32))) * D_IN + k];
    }
}

// A-fragment load: rows m -> Xpad[b][32 + t0 + m - off], 16-bit A 16x32 layout
__device__ __forceinline__ v16bf loadA(const __bf16* __restrict__ xp,
                                       int b, int t0, int off, int k0,
                                       int ml, int dlt)
{
    const int up = 32 + t0 + ml - off;           // <32 lands in the zero block
    const v8bf* ap = (const v8bf*)
        (xp + ((size_t)(b * XROWS + up)) * D_IN + k0 + dlt);
    v8bf a0 = ap[0];                             // K = k0+dlt    .. +7
    v8bf a1 = ap[1];                             // K = k0+8+dlt  .. +15
    v16bf a;
    #pragma unroll
    for (int e = 0; e < 8; ++e) { a[e] = a0[e]; a[e + 8] = a1[e]; }
    return a;
}

__device__ __forceinline__ v16bf loadB(const __bf16* sB, int k0)
{
    return *(const v16bf*)(sB + k0);
}

// One K-step, phase A (both t-tiles): 8 WMMAs, 3-deep A pipeline.
__device__ __forceinline__ void kstepA(v8f (&acc)[8], const __bf16* __restrict__ xp,
                                       int bg, int off, int k0, int ml, int dlt,
                                       v16bf bfr)
{
    v16bf af[3];
    af[0] = loadA(xp, bg, 0,  off, k0, ml, dlt);
    af[1] = loadA(xp, bg, 16, off, k0, ml, dlt);
    #pragma unroll
    for (int i = 0; i < 8; ++i) {
        if (i < 6) {
            const int j = i + 2;
            af[j % 3] = loadA(xp, bg + (j >> 1), (j & 1) * 16, off, k0, ml, dlt);
        }
        acc[i] = __builtin_amdgcn_wmma_f32_16x16x32_bf16(
            false, af[i % 3], false, bfr, (short)0, acc[i], false, false);
    }
}

// One K-step, phase B (upper t-tile only): 4 WMMAs, 3-deep A pipeline.
__device__ __forceinline__ void kstepB(v8f (&acc)[8], const __bf16* __restrict__ xp,
                                       int bg, int off, int k0, int ml, int dlt,
                                       v16bf bfr)
{
    v16bf af[3];
    af[0] = loadA(xp, bg + 0, 16, off, k0, ml, dlt);
    af[1] = loadA(xp, bg + 1, 16, off, k0, ml, dlt);
    #pragma unroll
    for (int i = 0; i < 4; ++i) {
        if (i < 2)
            af[(i + 2) % 3] = loadA(xp, bg + i + 2, 16, off, k0, ml, dlt);
        acc[i * 2 + 1] = __builtin_amdgcn_wmma_f32_16x16x32_bf16(
            false, af[i % 3], false, bfr, (short)0, acc[i * 2 + 1], false, false);
    }
}

// ---------------------------------------------------------------------------
// Kernel 1: causal-conv encoder GEMM.
//   pre[b,t,n] = sum_{off<=t} x[b,t-off,:] . W_enc[off][:,n]  (+ b_enc[n])
// 128 workgroups, each owning a disjoint 64-column slice of D_SAE for ALL
// 256 rows: every W_enc byte is read from HBM exactly once. bf16 WMMA.
// Wave w: column tile (w&3), batch group (w>>2) -> 8 acc tiles / wave.
// Branch-free two-phase tap loop; A 3-deep and B 2-deep software pipelines.
// ---------------------------------------------------------------------------
__global__ __launch_bounds__(256) void sae_encode_wmma(
    const __bf16* __restrict__ xp,    // (B, 64, D_IN) zero-padded bf16
    const float* __restrict__ Wenc,   // (T, D_IN, D_SAE)
    const float* __restrict__ benc,   // (D_SAE)
    float* __restrict__ pre)          // (NROWS, D_SAE) workspace
{
    extern __shared__ __bf16 sW[];    // [NBLK][KPAD] transposed bf16 tile

    const int tid  = threadIdx.x;
    const int lane = tid & 31;
    const int wv   = tid >> 5;        // 8 waves
    const int ct   = wv & 3;          // column tile (16 wide) within 64
    const int bg   = (wv >> 2) * 4;   // batch group base: 0 or 4
    const int n0   = blockIdx.x * NBLK;
    const bool hi  = (lane >= 16);
    const int ml   = lane & 15;       // A-row / B-col inside the 16-wide tile
    const int dlt  = hi ? 16 : 0;     // K sub-offset for this lane half

    v8f acc[8];                       // [ (b-bg)*2 + ttile ]
    v8f vz = 0;
    #pragma unroll
    for (int i = 0; i < 8; ++i) acc[i] = vz;

    const __bf16* sB = sW + (size_t)(ct * 16 + ml) * KPAD + (hi ? 16 : 0);

    for (int off = 0; off < Tn; ++off) {
        // ---- stage W_enc[off][:, n0:n0+64] into LDS, transposed, bf16.
        // Two k-rows per pass -> packed cvt + 4B-aligned v2bf LDS stores.
        __syncthreads();
        const float* wg = Wenc + (size_t)off * D_IN * D_SAE + n0;
        for (int i = tid; i < (D_IN / 2) * (NBLK / 4); i += 256) {
            int k2 = (i >> 4) * 2;                 // even k row
            int n4 = (i & 15) * 4;                 // coalesced 16B per row
            v4f w0 = *(const v4f*)(wg + (size_t)(k2 + 0) * D_SAE + n4);
            v4f w1 = *(const v4f*)(wg + (size_t)(k2 + 1) * D_SAE + n4);
            #pragma unroll
            for (int j = 0; j < 4; ++j) {
                v2bf p;
                p[0] = (__bf16)w0[j];
                p[1] = (__bf16)w1[j];
                *(v2bf*)(sW + (size_t)(n4 + j) * KPAD + k2) = p;
            }
        }
        __syncthreads();

        // prefetch next tap's slice while we compute (global_prefetch_b8)
        if (off + 1 < Tn) {
            const float* nxt = Wenc + (size_t)(off + 1) * D_IN * D_SAE + n0;
            #pragma unroll
            for (int p = 0; p < 3; ++p) {
                int k = tid * 3 + p;
                if (k < D_IN) __builtin_prefetch(nxt + (size_t)k * D_SAE, 0, 1);
            }
        }

        if (off < 16) {
            // ---- phase A: both t-tiles live; K unrolled x2, B double-buffered
            v16bf bfr0 = loadB(sB, 0), bfr1;
            for (int k0 = 0; k0 < D_IN; k0 += 64) {
                bfr1 = loadB(sB, k0 + 32);
                kstepA(acc, xp, bg, off, k0, ml, dlt, bfr0);
                if (k0 + 64 < D_IN) bfr0 = loadB(sB, k0 + 64);
                kstepA(acc, xp, bg, off, k0 + 32, ml, dlt, bfr1);
            }
        } else {
            // ---- phase B: only the upper t-tile (t0=16)
            v16bf bfr0 = loadB(sB, 0), bfr1;
            for (int k0 = 0; k0 < D_IN; k0 += 64) {
                bfr1 = loadB(sB, k0 + 32);
                kstepB(acc, xp, bg, off, k0, ml, dlt, bfr0);
                if (k0 + 64 < D_IN) bfr0 = loadB(sB, k0 + 64);
                kstepB(acc, xp, bg, off, k0 + 32, ml, dlt, bfr1);
            }
        }
    }

    // ---- write pre (+ b_enc) ----
    const int N = n0 + ct * 16 + ml;
    const float be = benc[N];
    #pragma unroll
    for (int i = 0; i < 8; ++i) {
        const int b  = bg + (i >> 1);
        const int t0 = (i & 1) * 16;
        #pragma unroll
        for (int r = 0; r < 8; ++r) {
            const int M   = r + (hi ? 8 : 0);     // C layout: vgpr r, lane half
            const int row = b * Tn + t0 + M;
            pre[(size_t)row * D_SAE + N] = acc[i][r] + be;
        }
    }
}

// ---------------------------------------------------------------------------
// Block all-reduce helpers (256 threads = 8 waves of 32)
// ---------------------------------------------------------------------------
__device__ inline float blkSumF(float v, volatile float* sb) {
    #pragma unroll
    for (int o = 16; o > 0; o >>= 1) v += __shfl_xor(v, o, 32);
    __syncthreads();
    if ((threadIdx.x & 31) == 0) sb[threadIdx.x >> 5] = v;
    __syncthreads();
    float s = 0.f;
    #pragma unroll
    for (int i = 0; i < 8; ++i) s += sb[i];
    return s;
}
__device__ inline float blkMaxF(float v, volatile float* sb) {
    #pragma unroll
    for (int o = 16; o > 0; o >>= 1) v = fmaxf(v, __shfl_xor(v, o, 32));
    __syncthreads();
    if ((threadIdx.x & 31) == 0) sb[threadIdx.x >> 5] = v;
    __syncthreads();
    float s = -3.4e38f;
    #pragma unroll
    for (int i = 0; i < 8; ++i) s = fmaxf(s, sb[i]);
    return s;
}
__device__ inline int blkSumI(int v, volatile int* sb) {
    #pragma unroll
    for (int o = 16; o > 0; o >>= 1) v += __shfl_xor(v, o, 32);
    __syncthreads();
    if ((threadIdx.x & 31) == 0) sb[threadIdx.x >> 5] = v;
    __syncthreads();
    int s = 0;
    #pragma unroll
    for (int i = 0; i < 8; ++i) s += sb[i];
    return s;
}

// ---------------------------------------------------------------------------
// Kernel 2: per-row Top-K(64) threshold by bisection, z = relu(pre) masked.
// ---------------------------------------------------------------------------
__global__ __launch_bounds__(256) void sae_topk(
    const float* __restrict__ pre, float* __restrict__ z)
{
    __shared__ float sf[8];
    __shared__ int   si[8];
    const int row = blockIdx.x, tid = threadIdx.x;
    const float* p = pre + (size_t)row * D_SAE;

    float v[32];
    float mx = -3.4e38f, mn = 3.4e38f;
    #pragma unroll
    for (int j = 0; j < 32; ++j) {
        v[j] = p[tid + 256 * j];
        mx = fmaxf(mx, v[j]);
        mn = fminf(mn, v[j]);
    }
    mx = blkMaxF(mx, sf);
    mn = -blkMaxF(-mn, sf);

    float lo = mn - 1.0f, hic = mx;   // invariant: count(>lo) >= K, count(>hi) < K
    for (int it = 0; it < 30; ++it) {
        float mid = 0.5f * (lo + hic);
        int c = 0;
        #pragma unroll
        for (int j = 0; j < 32; ++j) c += (v[j] > mid) ? 1 : 0;
        c = blkSumI(c, si);
        if (c >= TOPK) lo = mid; else hic = mid;
    }
    const float thr = lo;

    float* zr = z + (size_t)row * D_SAE;
    #pragma unroll
    for (int j = 0; j < 32; ++j) {
        float val = v[j];
        zr[tid + 256 * j] = (val > thr && val > 0.f) ? val : 0.f;
    }
}

// ---------------------------------------------------------------------------
// Kernel 3: sparse decode + loss.  x_hat[row] = b_dec + sum_nz z*W_dec[idx].
// ---------------------------------------------------------------------------
#define NZCAP 160
__global__ __launch_bounds__(256) void sae_decode(
    const float* __restrict__ z, const float* __restrict__ Wdec,
    const float* __restrict__ bdec, const float* __restrict__ x,
    float* __restrict__ xhat, float* __restrict__ loss)
{
    __shared__ int   s_idx[NZCAP];
    __shared__ float s_val[NZCAP];
    __shared__ int   s_cnt;
    __shared__ float sf[8];
    const int row = blockIdx.x, tid = threadIdx.x;

    if (tid == 0) s_cnt = 0;
    __syncthreads();

    const float* zr = z + (size_t)row * D_SAE;
    for (int c = tid; c < D_SAE; c += 256) {
        float v = zr[c];
        if (v > 0.f) {
            int p = atomicAdd(&s_cnt, 1);
            if (p < NZCAP) { s_idx[p] = c; s_val[p] = v; }
        }
    }
    __syncthreads();
    const int nnz = min(s_cnt, NZCAP);

    float part = 0.f;
    #pragma unroll
    for (int dd = 0; dd < 3; ++dd) {              // 768 = 3 * 256
        const int d = tid + dd * 256;
        float a = bdec[d];
        for (int j = 0; j < nnz; ++j)
            a = fmaf(s_val[j], Wdec[(size_t)s_idx[j] * D_IN + d], a);
        xhat[(size_t)row * D_IN + d] = a;
        float df = a - x[(size_t)row * D_IN + d];
        part = fmaf(df, df, part);
    }
    part = blkSumF(part, sf);
    if (tid == 0) atomicAdd(loss, part * (1.0f / (float)NROWS));
}

// ---------------------------------------------------------------------------
extern "C" void kernel_launch(void* const* d_in, const int* in_sizes, int n_in,
                              void* d_out, int out_size, void* d_ws, size_t ws_size,
                              hipStream_t stream) {
    (void)in_sizes; (void)n_in; (void)out_size; (void)ws_size;
    const float* x    = (const float*)d_in[0];
    const float* Wenc = (const float*)d_in[1];
    const float* Wdec = (const float*)d_in[2];
    const float* benc = (const float*)d_in[3];
    const float* bdec = (const float*)d_in[4];

    float* loss = (float*)d_out;                          // [0]
    float* xhat = loss + 1;                               // (B,T,D_IN)
    float* z    = xhat + (size_t)NROWS * D_IN;            // (B,T,D_SAE)

    float*  pre  = (float*)d_ws;                          // 8 MB
    __bf16* xpad = (__bf16*)((char*)d_ws +
                             (size_t)NROWS * D_SAE * sizeof(float)); // 768 KB

    // 0) zero-padded bf16 X (8*64*768 elems, 4 per thread)
    sae_build_xpad<<<(Bn * XROWS * D_IN) / 1024, 256, 0, stream>>>(x, xpad);

    // 1) encoder GEMM: 128 column-slice workgroups, ~100KB dynamic LDS each
    const size_t smem = (size_t)NBLK * KPAD * sizeof(__bf16);
    sae_encode_wmma<<<D_SAE / NBLK, 256, smem, stream>>>(xpad, Wenc, benc, pre);

    // 2) per-row top-K + relu -> z
    sae_topk<<<NROWS, 256, 0, stream>>>(pre, z);

    // 3) zero loss scalar, then sparse decode + loss accumulation
    (void)hipMemsetAsync(loss, 0, sizeof(float), stream);
    sae_decode<<<NROWS, 256, 0, stream>>>(z, Wdec, bdec, x, xhat, loss);
}
